// PointAttention_29257317220731
// MI455X (gfx1250) — matrix operand order repaired and checked by reading, not compiled, not run on hardware
//
#include <hip/hip_runtime.h>
#include <hip/hip_bf16.h>

typedef __attribute__((ext_vector_type(16))) __bf16 v16bf;
typedef __attribute__((ext_vector_type(8)))  float  v8f;
typedef __attribute__((ext_vector_type(4)))  unsigned int u32x4;
typedef __bf16 bf16_t;
typedef unsigned short ushort_t;

#define TPB 256
#define NW  8

union ABFrag { v16bf v; u32x4 q[2]; };
union CFrag  { v8f v; float f[8]; };

__device__ __forceinline__ ushort_t f2bf_bits(float x) {   // for pack kernel only
    unsigned int b = __float_as_uint(x);
    unsigned int r = b + 0x7FFFu + ((b >> 16) & 1u);
    return (ushort_t)(r >> 16);
}

// A fragment (16x32 bf16) from row-major LDS, row stride `stride` elements.
// ISA layout: lane holds row lane%16; half 0 -> K {k0..k0+7, k0+16..k0+23},
// half 1 -> K {k0+8..k0+15, k0+24..k0+31}.
__device__ __forceinline__ v16bf load_a_lds(const bf16_t* base, int row, int stride,
                                            int k0, int lane) {
    int half = lane >> 4;
    const bf16_t* p = base + row * stride + k0 + half * 8;
    ABFrag u;
    u.q[0] = *(const u32x4*)(p);
    u.q[1] = *(const u32x4*)(p + 16);
    return u.v;
}

// B fragment from packed weights: 32 contiguous bytes per lane.
__device__ __forceinline__ v16bf load_b_glb(const ushort_t* Bp, int ct, int KT,
                                            int kt, int lane) {
    const ushort_t* p = Bp + ((((ct * KT) + kt) * 32 + lane) << 4);
    ABFrag u;
    u.q[0] = ((const u32x4*)p)[0];
    u.q[1] = ((const u32x4*)p)[1];
    return u.v;
}

__device__ __forceinline__ v8f wmma_bf16(v16bf a, v16bf b, v8f c) {
    return __builtin_amdgcn_wmma_f32_16x16x32_bf16(false, a, false, b, (short)0, c,
                                                   false, false);
}

// Repack W[K][128] (f32) into bf16 B-fragment tiles:
// dst[(((ct*KT + kt)*32 + lane)*16) + e] = bf16(W[kt*32 + (e>>3)*16 + (lane>>4)*8 + (e&7)][ct*16 + lane%16])
__global__ void pa_pack_b(const float* __restrict__ W, ushort_t* __restrict__ dst, int K) {
    int total = K * 128;
    int KT = K >> 5;
    for (int o = blockIdx.x * blockDim.x + threadIdx.x; o < total;
         o += gridDim.x * blockDim.x) {
        int e    = o & 15;
        int lane = (o >> 4) & 31;
        int t    = o >> 9;
        int kt   = t % KT;
        int ct   = t / KT;
        int n = ct * 16 + (lane & 15);
        int k = kt * 32 + ((e >> 3) << 4) + ((lane >> 4) << 3) + (e & 7);
        dst[o] = f2bf_bits(W[k * 128 + n]);
    }
}

__global__ __launch_bounds__(TPB, 4) void pa_main(
    const float* __restrict__ xyz, const float* __restrict__ feat,
    const float* __restrict__ aw1, const float* __restrict__ ab1, const float* __restrict__ ab2,
    const float* __restrict__ bw1, const float* __restrict__ bb1, const float* __restrict__ bb2,
    const float* __restrict__ dw1, const float* __restrict__ db1, const float* __restrict__ db2,
    const float* __restrict__ eb1, const float* __restrict__ eb2,
    const float* __restrict__ ob1, const float* __restrict__ ob2,
    const ushort_t* __restrict__ aw2p, const ushort_t* __restrict__ bw2p,
    const ushort_t* __restrict__ dw2p, const ushort_t* __restrict__ ew1p,
    const ushort_t* __restrict__ ew2p, const ushort_t* __restrict__ ow1p,
    const ushort_t* __restrict__ ow2p,
    float* __restrict__ out)
{
    __shared__ bf16_t bufH[128 * 128];    // 32 KB (h1/h2 activations)
    __shared__ bf16_t bufT[128 * 128];    // 32 KB (t activations)
    __shared__ bf16_t ldsF[32 * 64];      //  4 KB (features bf16)
    __shared__ float qv[32 * 128];        // 16 KB
    __shared__ float kv[32 * 128];        // 16 KB
    __shared__ float vv[32 * 128];        // 16 KB
    __shared__ float denom[32 * 128];     // 16 KB
    __shared__ float numv[32 * 128];      // 16 KB
    __shared__ float biasL[3 * 128];      // 1.5 KB (db2 | eb1 | eb2)
    __shared__ float Xs[32 * 4];          // 0.5 KB

    const int b    = blockIdx.x;
    const int tid  = threadIdx.x;
    const int lane = tid & 31;
    const int wave = tid >> 5;
    const int n    = lane & 15;
    const int half = lane >> 4;
    const float scale = 0.08838834764831845f;   // 1/sqrt(128)

    // ---- stage 0: stage inputs into LDS, zero denom ----
    for (int t = tid; t < 32 * 3; t += TPB) {
        int i = t / 3, c = t % 3;
        Xs[i * 4 + c] = xyz[(b * 32 + i) * 3 + c];
    }
    for (int t = tid; t < 32 * 64; t += TPB) ldsF[t] = (bf16_t)feat[b * 2048 + t];
    for (int t = tid; t < 32 * 128; t += TPB) denom[t] = 0.f;
    for (int t = tid; t < 128; t += TPB) {
        biasL[t]       = db2[t];
        biasL[128 + t] = eb1[t];
        biasL[256 + t] = eb2[t];
    }
    __syncthreads();

    // ---- stage A1 (VALU): q/k first layer (din=3) -> bufH rows 0..63 ----
    for (int t = tid; t < 2 * 32 * 128; t += TPB) {
        int which = t >> 12;              // 0 = alpha(q), 1 = beta(k)
        int i = (t >> 7) & 31, f = t & 127;
        const float* w1 = which ? bw1 : aw1;
        const float* b1 = which ? bb1 : ab1;
        float acc = b1[f];
        acc += Xs[i * 4 + 0] * w1[0 * 128 + f];
        acc += Xs[i * 4 + 1] * w1[1 * 128 + f];
        acc += Xs[i * 4 + 2] * w1[2 * 128 + f];
        bufH[(which << 12) + (i << 7) + f] = (bf16_t)fmaxf(acc, 0.f);
    }
    __syncthreads();

    // ---- stage A2 (WMMA): hv = relu(F @ ow1 + ob1) -> bufT rows 0..31 ----
    for (int job = wave; job < 16; job += NW) {
        int rt = job >> 3, ct = job & 7;
        CFrag c; c.v = (v8f){};
#pragma unroll
        for (int kt = 0; kt < 2; ++kt) {
            v16bf a = load_a_lds(ldsF, rt * 16 + n, 64, kt * 32, lane);
            v16bf w = load_b_glb(ow1p, ct, 2, kt, lane);
            c.v = wmma_bf16(a, w, c.v);
        }
        int f = ct * 16 + n;
        float bias = ob1[f];
#pragma unroll
        for (int v = 0; v < 8; ++v)
            bufT[(rt * 16 + v + half * 8) * 128 + f] = (bf16_t)fmaxf(c.f[v] + bias, 0.f);
    }
    __syncthreads();

    // ---- stage A3 (WMMA): q,k,v second layers -> qv/kv/vv (f32, LDS) ----
    for (int job = wave; job < 48; job += NW) {
        int which = job / 16;             // 0 q, 1 k, 2 v
        int sub = job % 16;
        int rt = sub >> 3, ct = sub & 7;
        const bf16_t*   A  = (which == 0) ? bufH : (which == 1) ? (bufH + 32 * 128) : bufT;
        const ushort_t* Bp = (which == 0) ? aw2p : (which == 1) ? bw2p : ow2p;
        const float*    b2 = (which == 0) ? ab2  : (which == 1) ? bb2  : ob2;
        float*        outp = (which == 0) ? qv   : (which == 1) ? kv   : vv;
        CFrag c; c.v = (v8f){};
#pragma unroll
        for (int kt = 0; kt < 4; ++kt) {
            v16bf a = load_a_lds(A, rt * 16 + n, 128, kt * 32, lane);
            v16bf w = load_b_glb(Bp, ct, 4, kt, lane);
            c.v = wmma_bf16(a, w, c.v);
        }
        int f = ct * 16 + n;
        float bias = b2[f];
#pragma unroll
        for (int v = 0; v < 8; ++v)
            outp[(rt * 16 + v + half * 8) * 128 + f] = c.f[v] + bias;
    }

    // ---- B1 per-thread invariants (f = tid & 127 constant across it-loop) ----
    const int   f1  = tid & 127;
    const float d10 = dw1[0 * 128 + f1];
    const float d11 = dw1[1 * 128 + f1];
    const float d12 = dw1[2 * 128 + f1];
    const float db1f = db1[f1];
    // Wave <-> row-tile mapping: this wave owns pair-rows [wave*16, wave*16+16)
    // of each iteration's 128-row slab. A fragments are loaded ONCE per stage
    // and reused across all 8 feature column-tiles (B streams from L2-resident
    // packed weights -> no redundant LDS traffic).
    const int arow = wave * 16 + n;
    __syncthreads();

    // ---- stage B: 8 iterations x 128 pair-rows ----
    for (int it = 0; it < 8; ++it) {
        const int rt = it * 8 + wave;     // this wave's row-tile
        const int i  = rt >> 1;
        const int jb = (rt & 1) << 4;

        // B1 (VALU): h1 = relu(|xi-xj| @ dw1 + db1) -> bufH (bf16)
        for (int t = tid; t < 128 * 128; t += TPB) {
            int lr = t >> 7;
            int rr = it * 8 + (lr >> 4);
            int ii = rr >> 1;
            int jj = ((rr & 1) << 4) + (lr & 15);
            float acc = db1f;
            acc += fabsf(Xs[ii * 4 + 0] - Xs[jj * 4 + 0]) * d10;
            acc += fabsf(Xs[ii * 4 + 1] - Xs[jj * 4 + 1]) * d11;
            acc += fabsf(Xs[ii * 4 + 2] - Xs[jj * 4 + 2]) * d12;
            bufH[(lr << 7) + f1] = (bf16_t)fmaxf(acc, 0.f);
        }
        __syncthreads();

        // B2 (WMMA): t = (h1@dw2 + db2 + q_i - k_j)*scale -> bufT (bf16)
        {
            v16bf A0 = load_a_lds(bufH, arow, 128, 0,  lane);
            v16bf A1 = load_a_lds(bufH, arow, 128, 32, lane);
            v16bf A2 = load_a_lds(bufH, arow, 128, 64, lane);
            v16bf A3 = load_a_lds(bufH, arow, 128, 96, lane);
#pragma unroll 2
            for (int ct = 0; ct < 8; ++ct) {
                CFrag c; c.v = (v8f){};
                c.v = wmma_bf16(A0, load_b_glb(dw2p, ct, 4, 0, lane), c.v);
                c.v = wmma_bf16(A1, load_b_glb(dw2p, ct, 4, 1, lane), c.v);
                c.v = wmma_bf16(A2, load_b_glb(dw2p, ct, 4, 2, lane), c.v);
                c.v = wmma_bf16(A3, load_b_glb(dw2p, ct, 4, 3, lane), c.v);
                int f = ct * 16 + n;
                float qb = qv[i * 128 + f] + biasL[f];
#pragma unroll
                for (int v = 0; v < 8; ++v) {
                    int m = v + half * 8;
                    float tv = (c.f[v] + qb - kv[(jb + m) * 128 + f]) * scale;
                    bufT[(wave * 16 + m) * 128 + f] = (bf16_t)tv;
                }
            }
        }
        __syncthreads();

        // B3 (WMMA): h2 = relu(t@ew1 + eb1) -> bufH (bf16)
        {
            v16bf A0 = load_a_lds(bufT, arow, 128, 0,  lane);
            v16bf A1 = load_a_lds(bufT, arow, 128, 32, lane);
            v16bf A2 = load_a_lds(bufT, arow, 128, 64, lane);
            v16bf A3 = load_a_lds(bufT, arow, 128, 96, lane);
#pragma unroll 2
            for (int ct = 0; ct < 8; ++ct) {
                CFrag c; c.v = (v8f){};
                c.v = wmma_bf16(A0, load_b_glb(ew1p, ct, 4, 0, lane), c.v);
                c.v = wmma_bf16(A1, load_b_glb(ew1p, ct, 4, 1, lane), c.v);
                c.v = wmma_bf16(A2, load_b_glb(ew1p, ct, 4, 2, lane), c.v);
                c.v = wmma_bf16(A3, load_b_glb(ew1p, ct, 4, 3, lane), c.v);
                int f = ct * 16 + n;
                float bias = biasL[128 + f];
#pragma unroll
                for (int v = 0; v < 8; ++v)
                    bufH[(wave * 16 + v + half * 8) * 128 + f] =
                        (bf16_t)fmaxf(c.f[v] + bias, 0.f);
            }
        }
        __syncthreads();

        // B4 (WMMA): w = h2@ew2 + eb2; e = exp(w); accumulate denom, capture diag
        {
            v16bf A0 = load_a_lds(bufH, arow, 128, 0,  lane);
            v16bf A1 = load_a_lds(bufH, arow, 128, 32, lane);
            v16bf A2 = load_a_lds(bufH, arow, 128, 64, lane);
            v16bf A3 = load_a_lds(bufH, arow, 128, 96, lane);
#pragma unroll 2
            for (int ct = 0; ct < 8; ++ct) {
                CFrag c; c.v = (v8f){};
                c.v = wmma_bf16(A0, load_b_glb(ew2p, ct, 4, 0, lane), c.v);
                c.v = wmma_bf16(A1, load_b_glb(ew2p, ct, 4, 1, lane), c.v);
                c.v = wmma_bf16(A2, load_b_glb(ew2p, ct, 4, 2, lane), c.v);
                c.v = wmma_bf16(A3, load_b_glb(ew2p, ct, 4, 3, lane), c.v);
                int f = ct * 16 + n;
                float bias = biasL[256 + f];
                float s = 0.f;
#pragma unroll
                for (int v = 0; v < 8; ++v) {
                    int m = v + half * 8;
                    float e = __expf(c.f[v] + bias);
                    s += e;
                    if (jb + m == i) numv[i * 128 + f] = e;   // diagonal numerator
                }
                atomicAdd(&denom[i * 128 + f], s);            // ds_add_f32
            }
        }
        __syncthreads();
    }

    // ---- epilogue: res = softmax_diag * v ; concat features ----
    for (int t = tid; t < 32 * 192; t += TPB) {
        int i = t / 192, c = t % 192;
        float o;
        if (c < 128) o = numv[i * 128 + c] / denom[i * 128 + c] * vv[i * 128 + c];
        else         o = feat[b * 2048 + i * 64 + (c - 128)];
        out[b * 6144 + t] = o;
    }
}

extern "C" void kernel_launch(void* const* d_in, const int* in_sizes, int n_in,
                              void* d_out, int out_size, void* d_ws, size_t ws_size,
                              hipStream_t stream) {
    const float* xyz  = (const float*)d_in[0];
    const float* feat = (const float*)d_in[1];
    const float* aw1 = (const float*)d_in[2];  const float* ab1 = (const float*)d_in[3];
    const float* aw2 = (const float*)d_in[4];  const float* ab2 = (const float*)d_in[5];
    const float* bw1 = (const float*)d_in[6];  const float* bb1 = (const float*)d_in[7];
    const float* bw2 = (const float*)d_in[8];  const float* bb2 = (const float*)d_in[9];
    const float* dw1 = (const float*)d_in[10]; const float* db1 = (const float*)d_in[11];
    const float* dw2 = (const float*)d_in[12]; const float* db2 = (const float*)d_in[13];
    const float* ew1 = (const float*)d_in[14]; const float* eb1 = (const float*)d_in[15];
    const float* ew2 = (const float*)d_in[16]; const float* eb2 = (const float*)d_in[17];
    const float* ow1 = (const float*)d_in[18]; const float* ob1 = (const float*)d_in[19];
    const float* ow2 = (const float*)d_in[20]; const float* ob2 = (const float*)d_in[21];

    ushort_t* ws   = (ushort_t*)d_ws;
    ushort_t* aw2p = ws;
    ushort_t* bw2p = aw2p + 16384;
    ushort_t* dw2p = bw2p + 16384;
    ushort_t* ew1p = dw2p + 16384;
    ushort_t* ew2p = ew1p + 16384;
    ushort_t* ow2p = ew2p + 16384;
    ushort_t* ow1p = ow2p + 16384;   // K=64, 8192 elems

    pa_pack_b<<<64, TPB, 0, stream>>>(aw2, aw2p, 128);
    pa_pack_b<<<64, TPB, 0, stream>>>(bw2, bw2p, 128);
    pa_pack_b<<<64, TPB, 0, stream>>>(dw2, dw2p, 128);
    pa_pack_b<<<64, TPB, 0, stream>>>(ew1, ew1p, 128);
    pa_pack_b<<<64, TPB, 0, stream>>>(ew2, ew2p, 128);
    pa_pack_b<<<64, TPB, 0, stream>>>(ow2, ow2p, 128);
    pa_pack_b<<<32, TPB, 0, stream>>>(ow1, ow1p, 64);

    pa_main<<<1024, TPB, 0, stream>>>(xyz, feat,
                                      aw1, ab1, ab2, bw1, bb1, bb2,
                                      dw1, db1, db2, eb1, eb2, ob1, ob2,
                                      aw2p, bw2p, dw2p, ew1p, ew2p, ow1p, ow2p,
                                      (float*)d_out);
}
